// Head_44263932952582
// MI455X (gfx1250) — matrix operand order repaired and checked by reading, not compiled, not run on hardware
//
#include <hip/hip_runtime.h>

// ---------------------------------------------------------------------------
// Single attention head (B=16, T=2048, C=1024, H=128) for gfx1250 (MI455X).
// bf16 WMMA everywhere, fp32 accumulation, flash-attention formulation.
// ---------------------------------------------------------------------------

#define B_ 16
#define T_ 2048
#define C_ 1024
#define H_ 128
#define RB 64          // rows per block in the projection kernel

typedef __attribute__((ext_vector_type(16))) __bf16 v16bf;
typedef __attribute__((ext_vector_type(8)))  float  v8f;
typedef __attribute__((ext_vector_type(8)))  unsigned short us8;

union BFfrag { v16bf v; us8 h[2]; };

static __device__ inline v16bf mkfrag(us8 lo, us8 hi) {
    BFfrag u; u.h[0] = lo; u.h[1] = hi; return u.v;
}

static __device__ inline unsigned short f32_bf16(float f) {
    unsigned int u = __float_as_uint(f);
    u += 0x7FFFu + ((u >> 16) & 1u);           // round-to-nearest-even
    return (unsigned short)(u >> 16);
}

static __device__ inline v8f vzero() {
    v8f z = {0.f, 0.f, 0.f, 0.f, 0.f, 0.f, 0.f, 0.f};
    return z;
}

static __device__ inline v8f wmma_bf16(v16bf a, v16bf b, v8f c) {
    return __builtin_amdgcn_wmma_f32_16x16x32_bf16(
        /*neg_a=*/false, a, /*neg_b=*/false, b,
        /*c_mod=*/(short)0, c, /*reuse_a=*/false, /*reuse_b=*/false);
}

// ---------------------------------------------------------------------------
// Kernel 1: fused QKV projection, 64 rows per block.
//   Per 32-wide k-chunk: stage x chunk (64x32) + all three transposed weight
//   chunks (3 x 128x32) in LDS, then each wave (one 16-wide H tile) does
//   12 WMMAs (4 M-tiles x {q,k,v}), reusing 3 B-fragments.
//   q stored pre-scaled by sqrt(C)=32; v stored transposed [H][T].
// ---------------------------------------------------------------------------
__global__ __launch_bounds__(256)
void qkv_proj_kernel(const float* __restrict__ x,
                     const float* __restrict__ Wq, const float* __restrict__ bq,
                     const float* __restrict__ Wk, const float* __restrict__ bk,
                     const float* __restrict__ Wv, const float* __restrict__ bv,
                     unsigned short* __restrict__ Qb,
                     unsigned short* __restrict__ Kb,
                     unsigned short* __restrict__ Vt)
{
    __shared__ unsigned short xs[RB * 32];          //  4 KB: x chunk [row][kk]
    __shared__ unsigned short wsh[3 * 128 * 32];    // 24 KB: Wt[mat][n][kk]

    const int tid  = threadIdx.x;
    const int lane = tid & 31;
    const int w    = tid >> 5;
    const int hi   = lane >> 4;      // lane half
    const int lm   = lane & 15;
    const int rowbase = blockIdx.x * RB;            // flat (b*T + t) row base

    // Fixed staging geometry (constant trip counts, induction addressing).
    const int xc  = tid & 31;        // k-within-chunk for x staging
    const int xr0 = tid >> 5;        // row start (stride 8)
    const int wn  = tid & 127;       // n for weight staging (coalesced)
    const int wk0 = tid >> 7;        // kk start (stride 2)

    const int n0 = w * 16;           // this wave's H tile
    v8f acc[4][3];
#pragma unroll
    for (int mt = 0; mt < 4; ++mt)
#pragma unroll
        for (int mat = 0; mat < 3; ++mat) acc[mt][mat] = vzero();

    for (int kc = 0; kc < C_; kc += 32) {
        __syncthreads();
        // Stage x chunk as bf16: xs[r][kk] = x[rowbase+r][kc+kk]
#pragma unroll
        for (int j = 0; j < 8; ++j) {
            int r = xr0 + 8 * j;
            xs[r * 32 + xc] = f32_bf16(x[(size_t)(rowbase + r) * C_ + kc + xc]);
        }
        // Stage transposed weight chunks: wsh[mat][n][kk] = W[(kc+kk)*H + n]
#pragma unroll
        for (int j = 0; j < 16; ++j) {
            int kk = wk0 + 2 * j;
            size_t g = (size_t)(kc + kk) * H_ + wn;
            int   l  = wn * 32 + kk;
            wsh[0 * 4096 + l] = f32_bf16(Wq[g]);
            wsh[1 * 4096 + l] = f32_bf16(Wk[g]);
            wsh[2 * 4096 + l] = f32_bf16(Wv[g]);
        }
        __syncthreads();

        // B fragments (32x16) for this wave's n-tile, one per matrix.
        v16bf fb[3];
#pragma unroll
        for (int mat = 0; mat < 3; ++mat) {
            const unsigned short* p = &wsh[mat * 4096 + (n0 + lm) * 32 + hi * 8];
            fb[mat] = mkfrag(*(const us8*)p, *(const us8*)(p + 16));
        }
        // 4 A fragments (16x32) from the x chunk; 12 WMMAs total.
#pragma unroll
        for (int mt = 0; mt < 4; ++mt) {
            const unsigned short* ar = &xs[(mt * 16 + lm) * 32 + hi * 8];
            v16bf a = mkfrag(*(const us8*)ar, *(const us8*)(ar + 16));
            acc[mt][0] = wmma_bf16(a, fb[0], acc[mt][0]);
            acc[mt][1] = wmma_bf16(a, fb[1], acc[mt][1]);
            acc[mt][2] = wmma_bf16(a, fb[2], acc[mt][2]);
        }
    }

    // Bias, scale (q only), store.
    const int ncol = n0 + lm;
    const float vbq = bq[ncol], vbk = bk[ncol], vbv = bv[ncol];
    const int bidx = rowbase / T_;                  // block never straddles batches
    const int t0   = rowbase % T_;

#pragma unroll
    for (int mt = 0; mt < 4; ++mt) {
        us8 vpack;
#pragma unroll
        for (int r = 0; r < 8; ++r) {
            int row = rowbase + mt * 16 + r + hi * 8;   // flat row
            Qb[(size_t)row * H_ + ncol] = f32_bf16((acc[mt][0][r] + vbq) * 32.0f);
            Kb[(size_t)row * H_ + ncol] = f32_bf16(acc[mt][1][r] + vbk);
            vpack[r] = f32_bf16(acc[mt][2][r] + vbv);
        }
        // V transposed: Vt[b][h][t]; rows r map to consecutive t -> b128 store.
        *(us8*)&Vt[((size_t)bidx * H_ + ncol) * T_ + t0 + mt * 16 + hi * 8] = vpack;
    }
}

// ---------------------------------------------------------------------------
// Kernel 2: causal flash attention. One wave per 16-query tile, key chunk 32.
// ---------------------------------------------------------------------------
__global__ __launch_bounds__(256)
void flash_attn_kernel(const unsigned short* __restrict__ Qb,
                       const unsigned short* __restrict__ Kb,
                       const unsigned short* __restrict__ Vt,
                       float* __restrict__ out)
{
    __shared__ unsigned short ps[8][16 * 32];       // per-wave P staging (8 KB)

    const int tid  = threadIdx.x;
    const int lane = tid & 31;
    const int w    = tid >> 5;
    const int hi   = lane >> 4;
    const int lm   = lane & 15;
    const int b    = blockIdx.y;
    const int qbase = blockIdx.x * 128 + w * 16;

    const unsigned short* Q = Qb + (size_t)b * T_ * H_;
    const unsigned short* K = Kb + (size_t)b * T_ * H_;
    const unsigned short* V = Vt + (size_t)b * H_ * T_;   // [H][T]

    // Q fragments: 4 chunks of 32 along H, contiguous b128 loads.
    v16bf qf[4];
#pragma unroll
    for (int c = 0; c < 4; ++c) {
        const unsigned short* ar = &Q[(size_t)(qbase + lm) * H_ + 32 * c + hi * 8];
        qf[c] = mkfrag(*(const us8*)ar, *(const us8*)(ar + 16));
    }

    v8f o[8];
    float mi[8], li[8];
#pragma unroll
    for (int nt = 0; nt < 8; ++nt) o[nt] = vzero();
#pragma unroll
    for (int r = 0; r < 8; ++r) { mi[r] = -__builtin_inff(); li[r] = 0.f; }

    const int qend = qbase + 16;
    for (int kb = 0; kb < qend; kb += 32) {
        if (kb + 32 < qend)
            __builtin_prefetch(&K[(size_t)(kb + 32 + lm) * H_], 0, 3);

        // S = Q * K^T for keys [kb, kb+32): two 16x16 accumulators.
        v8f s0 = vzero(), s1 = vzero();
#pragma unroll
        for (int c = 0; c < 4; ++c) {
            const unsigned short* k0 = &K[(size_t)(kb + lm) * H_ + 32 * c + hi * 8];
            const unsigned short* k1 = &K[(size_t)(kb + 16 + lm) * H_ + 32 * c + hi * 8];
            v16bf fb0 = mkfrag(*(const us8*)k0, *(const us8*)(k0 + 16));
            v16bf fb1 = mkfrag(*(const us8*)k1, *(const us8*)(k1 + 16));
            s0 = wmma_bf16(qf[c], fb0, s0);
            s1 = wmma_bf16(qf[c], fb1, s1);
        }

        // Causal mask in C layout: VGPR r holds row (r+8*hi), lane lm = col.
#pragma unroll
        for (int r = 0; r < 8; ++r) {
            int row = qbase + r + hi * 8;
            if (kb + lm > row)      s0[r] = -__builtin_inff();
            if (kb + 16 + lm > row) s1[r] = -__builtin_inff();
        }

        // Online softmax: row max / running stats via 16-lane xor reductions.
        float mnew[8], alpha[8];
#pragma unroll
        for (int r = 0; r < 8; ++r) {
            float m = fmaxf(s0[r], s1[r]);
            m = fmaxf(m, __shfl_xor(m, 1));
            m = fmaxf(m, __shfl_xor(m, 2));
            m = fmaxf(m, __shfl_xor(m, 4));
            m = fmaxf(m, __shfl_xor(m, 8));
            mnew[r]  = fmaxf(mi[r], m);
            alpha[r] = __expf(mi[r] - mnew[r]);
            mi[r]    = mnew[r];
        }

        // Exponentiate, accumulate row sums, spill P (bf16) to per-wave LDS.
#pragma unroll
        for (int r = 0; r < 8; ++r) {
            float p0 = __expf(s0[r] - mnew[r]);
            float p1 = __expf(s1[r] - mnew[r]);
            float ls = p0 + p1;
            ls += __shfl_xor(ls, 1);
            ls += __shfl_xor(ls, 2);
            ls += __shfl_xor(ls, 4);
            ls += __shfl_xor(ls, 8);
            li[r] = li[r] * alpha[r] + ls;
            ps[w][(r + hi * 8) * 32 + lm]      = f32_bf16(p0);
            ps[w][(r + hi * 8) * 32 + 16 + lm] = f32_bf16(p1);
        }

        // Rescale running output.
#pragma unroll
        for (int nt = 0; nt < 8; ++nt)
#pragma unroll
            for (int r = 0; r < 8; ++r) o[nt][r] *= alpha[r];

        __builtin_amdgcn_wave_barrier();
        // Reload P as A-fragment (wave-private LDS, DS ops are in-order).
        const unsigned short* pr = &ps[w][lm * 32 + hi * 8];
        v16bf pf = mkfrag(*(const us8*)pr, *(const us8*)(pr + 16));
        __builtin_amdgcn_wave_barrier();

        // O += P * V; V is [H][T] so B-fragments are contiguous b128 loads.
#pragma unroll
        for (int nt = 0; nt < 8; ++nt) {
            const unsigned short* vp = &V[(size_t)(nt * 16 + lm) * T_ + kb + hi * 8];
            v16bf vf = mkfrag(*(const us8*)vp, *(const us8*)(vp + 16));
            o[nt] = wmma_bf16(pf, vf, o[nt]);
        }
    }

    // Normalize and store fp32 output [B][T][H].
#pragma unroll
    for (int r = 0; r < 8; ++r) li[r] = 1.0f / li[r];
    float* op = out + ((size_t)b * T_ + qbase) * H_;
#pragma unroll
    for (int nt = 0; nt < 8; ++nt)
#pragma unroll
        for (int r = 0; r < 8; ++r)
            op[(size_t)(r + hi * 8) * H_ + nt * 16 + lm] = o[nt][r] * li[r];
}

// ---------------------------------------------------------------------------
// Launch. Workspace layout (ushort elements):
//   Qb : B*T*H                    (bf16, pre-scaled by 32)
//   Kb : B*T*H + 16*H pad         (pad absorbs tail-chunk overreads)
//   Vt : B*H*T + 32 pad           (transposed V)
// Total ~25.2 MB.
// ---------------------------------------------------------------------------
extern "C" void kernel_launch(void* const* d_in, const int* in_sizes, int n_in,
                              void* d_out, int out_size, void* d_ws, size_t ws_size,
                              hipStream_t stream) {
    const float* x  = (const float*)d_in[0];
    const float* Wq = (const float*)d_in[1];
    const float* bq = (const float*)d_in[2];
    const float* Wk = (const float*)d_in[3];
    const float* bk = (const float*)d_in[4];
    const float* Wv = (const float*)d_in[5];
    const float* bv = (const float*)d_in[6];
    float* out = (float*)d_out;

    const size_t BTH = (size_t)B_ * T_ * H_;
    unsigned short* Qb = (unsigned short*)d_ws;
    unsigned short* Kb = Qb + BTH;
    unsigned short* Vt = Kb + BTH + 16 * H_;   // pad after K

    qkv_proj_kernel<<<dim3((B_ * T_) / RB), dim3(256), 0, stream>>>(
        x, Wq, bq, Wk, bk, Wv, bv, Qb, Kb, Vt);

    flash_attn_kernel<<<dim3(T_ / 128, B_), dim3(256), 0, stream>>>(
        Qb, Kb, Vt, out);
}
